// MarketStateManager_45217415692845
// MI455X (gfx1250) — matrix operand order repaired and checked by reading, not compiled, not run on hardware
//
#include <hip/hip_runtime.h>
#include <hip/hip_bf16.h>

typedef __attribute__((ext_vector_type(16))) _Float16 v16h;
typedef __attribute__((ext_vector_type(8)))  _Float16 v8h;
typedef __attribute__((ext_vector_type(8)))  float    v8f;
typedef __attribute__((ext_vector_type(4)))  float    v4f;
typedef __attribute__((ext_vector_type(4)))  unsigned int u32x4;
typedef __attribute__((ext_vector_type(4)))  int          i32x4;
typedef __attribute__((ext_vector_type(8)))  int          i32x8;

#define D_MODEL 64
#define S_DIM   32
#define BATCH   256
#define SEQ_T   2048

#if __has_builtin(__builtin_amdgcn_tensor_load_to_lds)
#define HAVE_TDM 1
#else
#define HAVE_TDM 0
#endif

__device__ __forceinline__ float sigmoidf_(float x) {
    return 1.0f / (1.0f + __expf(-x));
}

template <int N>
__device__ __forceinline__ void wait_tensorcnt() {
#if __has_builtin(__builtin_amdgcn_s_wait_tensorcnt)
    __builtin_amdgcn_s_wait_tensorcnt(N);
#else
    if (N == 0) asm volatile("s_wait_tensorcnt 0x0" ::: "memory");
    else        asm volatile("s_wait_tensorcnt 0x1" ::: "memory");
#endif
    asm volatile("" ::: "memory");   // compiler barrier: TDM wrote LDS behind its back
}

#if HAVE_TDM
// TDM: DMA one 16x64 f32 tile (row stride T*D elements) from global into LDS.
// D# per CDNA5 ISA: 2D tile, data_size=4B, groups 2/3 zero.
__device__ __forceinline__ void tdm_load_x_tile(const float* gsrc, unsigned lds_off) {
    unsigned long long ga = (unsigned long long)(size_t)gsrc;

    u32x4 g0;
    g0[0] = 1u;                                               // count=1 (valid user D#)
    g0[1] = lds_off;                                          // lds_addr (bytes)
    g0[2] = (unsigned)(ga & 0xFFFFFFFFu);                     // global_addr[31:0]
    g0[3] = (unsigned)((ga >> 32) & 0x01FFFFFFu) | (2u << 30); // addr[56:32] | type=2

    i32x8 g1;
    g1[0] = 0x00020000;                   // workgroup_mask=0, data_size=2 (4 bytes)
    g1[1] = (int)(64u << 16);             // tensor_dim0 = 64   (bits 79:48, low half)
    g1[2] = (int)(16u << 16);             // tensor_dim1 = 16   (bits 111:80, low half)
    g1[3] = (int)(64u << 16);             // tile_dim0   = 64   (bits 127:112)
    g1[4] = 16;                           // tile_dim1   = 16   (bits 143:128), tile_dim2=0
    g1[5] = (int)(SEQ_T * D_MODEL);       // tensor_dim0_stride = 131072 (bits 207:160)
    g1[6] = 0;                            // stride hi, dim1_stride lo
    g1[7] = 0;                            // dim1_stride hi (unused for 2D)

    i32x4 gz = {0, 0, 0, 0};
#if defined(__clang_major__) && __clang_major__ >= 23
    i32x8 gz8 = {0, 0, 0, 0, 0, 0, 0, 0};
    __builtin_amdgcn_tensor_load_to_lds(g0, g1, gz, gz, gz8, 0);
#else
    __builtin_amdgcn_tensor_load_to_lds(g0, g1, gz, gz, 0);
#endif
}
#endif

// Build a v16h fragment from two contiguous 16-byte f16 runs in LDS.
__device__ __forceinline__ v16h ld_frag(const _Float16* p0, const _Float16* p1) {
    v8h a = *(const v8h*)p0;
    v8h b = *(const v8h*)p1;
    v16h f;
#pragma unroll
    for (int i = 0; i < 8; ++i) { f[i] = a[i]; f[i + 8] = b[i]; }
    return f;
}

// Build a v16h A-fragment from two contiguous 32-byte f32 runs in LDS (cvt to f16).
__device__ __forceinline__ v16h ld_frag_f32(const float* p0, const float* p1) {
    v4f a0 = *(const v4f*)p0;
    v4f a1 = *(const v4f*)(p0 + 4);
    v4f b0 = *(const v4f*)p1;
    v4f b1 = *(const v4f*)(p1 + 4);
    v16h f;
#pragma unroll
    for (int i = 0; i < 4; ++i) {
        f[i]      = (_Float16)a0[i];
        f[i + 4]  = (_Float16)a1[i];
        f[i + 8]  = (_Float16)b0[i];
        f[i + 12] = (_Float16)b1[i];
    }
    return f;
}

// B fragment: weights stored [N][Kstride] row-major in LDS (f16).
// Lane l holds N = ntile*16 + (l&15); element j holds K = koff + 8*(l>>4) + j (j<8), koff + 16 + 8*(l>>4) + (j-8) (j>=8).
__device__ __forceinline__ v16h ld_bfrag(const _Float16* base, int ntile, int koff,
                                         int kstride, int lo, int hi) {
    const _Float16* p = base + (size_t)(ntile * 16 + lo) * kstride + koff + 8 * hi;
    return ld_frag(p, p + 16);
}

#define WMMA(A, B, C) \
    __builtin_amdgcn_wmma_f32_16x16x32_f16(false, (A), false, (B), (short)0, (C), false, false)

__global__ void __launch_bounds__(32)
msm_scan_kernel(const float* __restrict__ trade_feat,   // (B,T,64)
                const float* __restrict__ has_trade,    // (B,T)
                const float* __restrict__ init_state,   // (1,32)
                const float* __restrict__ w_update,     // (32,96)
                const float* __restrict__ b_update,     // (32)
                const float* __restrict__ w_reset,      // (32,96)
                const float* __restrict__ b_reset,      // (32)
                const float* __restrict__ w_ih,         // (96,64)
                const float* __restrict__ w_hh,         // (96,32)
                const float* __restrict__ b_ih,         // (96)
                const float* __restrict__ b_hh,         // (96)
                const float* __restrict__ decay_rate,   // (1)
                float* __restrict__ out)                // (B,T,32)
{
    // Weights as f16 in LDS, [N][K] row-major (torch (out,in) layout is already this).
    __shared__ __align__(16) _Float16 s_wur[64 * 96];  // rows 0..31 w_update, 32..63 w_reset
    __shared__ __align__(16) _Float16 s_wih[96 * 64];
    __shared__ __align__(16) _Float16 s_whh[96 * 32];
    __shared__ __align__(16) float    s_xf[2][16 * 64]; // double-buffered x tile (f32, TDM dest)
    __shared__ __align__(16) _Float16 s_h[16 * 32];     // state, f16 [M][K]
    __shared__ __align__(16) _Float16 s_hr[16 * 32];    // state*rg, f16 [M][K]
    __shared__ float s_ev[16];

    const int lane = threadIdx.x;   // 0..31
    const int lo   = lane & 15;
    const int hi   = lane >> 4;
    const int b0   = blockIdx.x * 16;

    // ---- preload weights (f32 -> f16, contiguous copy) ----
    for (int i = lane; i < 32 * 96; i += 32) s_wur[i]           = (_Float16)w_update[i];
    for (int i = lane; i < 32 * 96; i += 32) s_wur[32 * 96 + i] = (_Float16)w_reset[i];
    for (int i = lane; i < 96 * 64; i += 32) s_wih[i]           = (_Float16)w_ih[i];
    for (int i = lane; i < 96 * 32; i += 32) s_whh[i]           = (_Float16)w_hh[i];

    // ---- per-lane biases at this lane's N positions ----
    const float bu0 = b_update[lo], bu1 = b_update[16 + lo];
    const float br0 = b_reset[lo],  br1 = b_reset[16 + lo];
    float bih[6], bhh[6];
#pragma unroll
    for (int n = 0; n < 6; ++n) { bih[n] = b_ih[n * 16 + lo]; bhh[n] = b_hh[n * 16 + lo]; }

    const float dec = sigmoidf_(decay_rate[0]);

    // state in WMMA C/D layout: lane l, reg r -> M = r + 8*hi, N = lo / 16+lo
    float hA[8], hB[8];
#pragma unroll
    for (int r = 0; r < 8; ++r) { hA[r] = init_state[lo]; hB[r] = init_state[16 + lo]; }

    const int am = lo;        // A-fragment row (M) for this lane
    const int ak = 8 * hi;    // A-fragment K sub-offset for this lane

    // ---- prologue: stage x tile for t=0 ----
#if HAVE_TDM
    const unsigned lds_x0 = (unsigned)(size_t)(void*)&s_xf[0][0];
    const unsigned lds_x1 = (unsigned)(size_t)(void*)&s_xf[1][0];
    tdm_load_x_tile(trade_feat + (size_t)b0 * SEQ_T * D_MODEL, lds_x0);
#else
    {
        const int m = lane >> 1, c = (lane & 1) * 32;
        const float* src = trade_feat + ((size_t)(b0 + m) * SEQ_T + 0) * D_MODEL + c;
#pragma unroll
        for (int i = 0; i < 8; ++i) *(v4f*)&s_xf[0][m * 64 + c + i * 4] = *(const v4f*)(src + i * 4);
    }
#endif
    __syncthreads();

    for (int t = 0; t < SEQ_T; ++t) {
        const float* xb = s_xf[t & 1];

        // ---- kick DMA of next x tile into the other buffer (overlaps this step) ----
        if (t + 1 < SEQ_T) {
#if HAVE_TDM
            tdm_load_x_tile(trade_feat + ((size_t)b0 * SEQ_T + (t + 1)) * D_MODEL,
                            ((t + 1) & 1) ? lds_x1 : lds_x0);
#else
            const int m = lane >> 1, c = (lane & 1) * 32;
            const float* src = trade_feat + ((size_t)(b0 + m) * SEQ_T + (t + 1)) * D_MODEL + c;
            float* dst = s_xf[(t + 1) & 1];
#pragma unroll
            for (int i = 0; i < 8; ++i) *(v4f*)&dst[m * 64 + c + i * 4] = *(const v4f*)(src + i * 4);
#endif
        }

        if (lane < 16) {
            s_ev[lane] = has_trade[(size_t)(b0 + lane) * SEQ_T + t];
            if (t + 1 < SEQ_T) __builtin_prefetch(&has_trade[(size_t)(b0 + lane) * SEQ_T + t + 1], 0, 0);
        }

        // ---- stage current state into LDS as f16 ----
#pragma unroll
        for (int r = 0; r < 8; ++r) {
            const int m = r + 8 * hi;
            s_h[m * 32 + lo]      = (_Float16)hA[r];
            s_h[m * 32 + 16 + lo] = (_Float16)hB[r];
        }

#if HAVE_TDM
        // retire the DMA that produced this step's buffer (one newer load may stay in flight)
        if (t + 1 < SEQ_T) wait_tensorcnt<1>();
        else               wait_tensorcnt<0>();
#endif
        __syncthreads();

        // ---- A fragments: combined = [x(K0..63) | h(K64..95)] as three K=32 chunks ----
        v16h Afr[3];
        Afr[0] = ld_frag_f32(&xb[am * 64 + 0  + ak], &xb[am * 64 + 16 + ak]);
        Afr[1] = ld_frag_f32(&xb[am * 64 + 32 + ak], &xb[am * 64 + 48 + ak]);
        Afr[2] = ld_frag(&s_h[am * 32 + ak], &s_h[am * 32 + 16 + ak]);

        // ---- ug/rg: combined(16x96) @ Wur^T(96x64) -> 4 N-tiles x 3 K-chunks ----
        v8f accUR[4] = {};
#pragma unroll
        for (int n = 0; n < 4; ++n)
#pragma unroll
            for (int kc = 0; kc < 3; ++kc)
                accUR[n] = WMMA(Afr[kc], ld_bfrag(s_wur, n, kc * 32, 96, lo, hi), accUR[n]);

        // ---- gi: x(16x64) @ w_ih^T(64x96) -> 6 N-tiles x 2 K-chunks ----
        v8f accI[6] = {};
#pragma unroll
        for (int n = 0; n < 6; ++n)
#pragma unroll
            for (int kc = 0; kc < 2; ++kc)
                accI[n] = WMMA(Afr[kc], ld_bfrag(s_wih, n, kc * 32, 64, lo, hi), accI[n]);

        // ---- gates ug, rg; stage hr = h*rg into LDS ----
        float ugA[8], ugB[8], hrA[8], hrB[8];
#pragma unroll
        for (int r = 0; r < 8; ++r) {
            ugA[r] = sigmoidf_(accUR[0][r] + bu0);
            ugB[r] = sigmoidf_(accUR[1][r] + bu1);
            const float rg0 = sigmoidf_(accUR[2][r] + br0);
            const float rg1 = sigmoidf_(accUR[3][r] + br1);
            hrA[r] = hA[r] * rg0;
            hrB[r] = hB[r] * rg1;
            const int m = r + 8 * hi;
            s_hr[m * 32 + lo]      = (_Float16)hrA[r];
            s_hr[m * 32 + 16 + lo] = (_Float16)hrB[r];
        }
        __syncthreads();

        // ---- gh: hr(16x32) @ w_hh^T(32x96) -> 6 N-tiles x 1 K-chunk ----
        const v16h Ahr = ld_frag(&s_hr[am * 32 + ak], &s_hr[am * 32 + 16 + ak]);
        v8f accG[6] = {};
#pragma unroll
        for (int n = 0; n < 6; ++n)
            accG[n] = WMMA(Ahr, ld_bfrag(s_whh, n, 0, 32, lo, hi), accG[n]);

        // ---- GRU combine + event gating + output store ----
#pragma unroll
        for (int r = 0; r < 8; ++r) {
            const int m  = r + 8 * hi;
            const float ev = s_ev[m];

            const float irA = accI[0][r] + bih[0], irB = accI[1][r] + bih[1];
            const float izA = accI[2][r] + bih[2], izB = accI[3][r] + bih[3];
            const float inA = accI[4][r] + bih[4], inB = accI[5][r] + bih[5];
            const float grA = accG[0][r] + bhh[0], grB = accG[1][r] + bhh[1];
            const float gzA = accG[2][r] + bhh[2], gzB = accG[3][r] + bhh[3];
            const float gnA = accG[4][r] + bhh[4], gnB = accG[5][r] + bhh[5];

            const float rA = sigmoidf_(irA + grA), rB = sigmoidf_(irB + grB);
            const float zA = sigmoidf_(izA + gzA), zB = sigmoidf_(izB + gzB);
            const float nA = tanhf(inA + rA * gnA), nB = tanhf(inB + rB * gnB);

            const float newA = (1.0f - zA) * nA + zA * hrA[r];
            const float newB = (1.0f - zB) * nB + zB * hrB[r];

            const float sA = (ev > 0.5f) ? (ugA[r] * newA + (1.0f - ugA[r]) * hA[r]) : dec * hA[r];
            const float sB = (ev > 0.5f) ? (ugB[r] * newB + (1.0f - ugB[r]) * hB[r]) : dec * hB[r];
            hA[r] = sA;
            hB[r] = sB;

            float* op = out + ((size_t)(b0 + m) * SEQ_T + t) * S_DIM;
            op[lo]      = sA;
            op[16 + lo] = sB;
        }
        __syncthreads();   // protect s_h / s_hr / s_ev before next step's staging
    }
}

extern "C" void kernel_launch(void* const* d_in, const int* in_sizes, int n_in,
                              void* d_out, int out_size, void* d_ws, size_t ws_size,
                              hipStream_t stream) {
    (void)in_sizes; (void)n_in; (void)out_size; (void)d_ws; (void)ws_size;
    const float* trade_feat = (const float*)d_in[0];
    const float* has_trade  = (const float*)d_in[1];
    const float* init_state = (const float*)d_in[2];
    const float* w_update   = (const float*)d_in[3];
    const float* b_update   = (const float*)d_in[4];
    const float* w_reset    = (const float*)d_in[5];
    const float* b_reset    = (const float*)d_in[6];
    const float* w_ih       = (const float*)d_in[7];
    const float* w_hh       = (const float*)d_in[8];
    const float* b_ih       = (const float*)d_in[9];
    const float* b_hh       = (const float*)d_in[10];
    const float* decay_rate = (const float*)d_in[11];
    float* out = (float*)d_out;

    dim3 grid(BATCH / 16);
    dim3 block(32);
    hipLaunchKernelGGL(msm_scan_kernel, grid, block, 0, stream,
                       trade_feat, has_trade, init_state,
                       w_update, b_update, w_reset, b_reset,
                       w_ih, w_hh, b_ih, b_hh, decay_rate, out);
}
